// SelfAdaptiveGPT2_58626303590711
// MI455X (gfx1250) — compile-verified, matrix operand order checked by reading, not run on hardware
//
#include <hip/hip_runtime.h>
#include <hip/hip_bf16.h>
#include <math.h>

typedef __attribute__((ext_vector_type(16))) __bf16 v16bf;
typedef __attribute__((ext_vector_type(8)))  __bf16 v8bf;
typedef __attribute__((ext_vector_type(8)))  float  v8f;

#define TM 128
#define TN 128
#define TK 32
#define LDSTR 40   // bf16 elems per LDS row (80B pitch, 16B aligned)

// dynamic LDS layout (bytes)
#define AS_OFF    0
#define BS_OFF    (TM * LDSTR * 2)                    // 10240
#define ASTG_OFF  (BS_OFF + TN * LDSTR * 2)           // 20480
#define BSTG_OFF  (ASTG_OFF + 2 * TM * TK * 4)        // 53248
#define SMEM_BYTES (BSTG_OFF + 2 * TN * TK * 4)       // 86016

union Frag { v16bf v; v8bf h[2]; };

__device__ __forceinline__ float gelu_tanh(float v) {
    return 0.5f * v * (1.0f + tanhf(0.7978845608028654f * (v + 0.044715f * v * v * v)));
}

// ---------------------------------------------------------------------------
// Async global->LDS copy of 16 bytes per lane (CDNA5 GLOBAL_LOAD_ASYNC_TO_LDS).
// LDS destination address = low 32 bits of the generic pointer (LDS aperture
// carries the LDS offset in addr[31:0]). Tracked by ASYNCcnt.
// ---------------------------------------------------------------------------
__device__ __forceinline__ void async_ld16(const float* g, float* l) {
    unsigned int       loff = (unsigned int)(uintptr_t)l;
    unsigned long long ga   = (unsigned long long)(uintptr_t)g;
    asm volatile("global_load_async_to_lds_b128 %0, %1, off"
                 :: "v"(loff), "v"(ga) : "memory");
}
__device__ __forceinline__ void async_wait8() {
    asm volatile("s_wait_asynccnt 0x8" ::: "memory");
}
__device__ __forceinline__ void async_wait0() {
    asm volatile("s_wait_asynccnt 0x0" ::: "memory");
}

// ---------------------------------------------------------------------------
// Issue one K-tile of async copies into fp32 staging buffers.
// Exactly 8 async ops per wave (4 for A's 128x32, 4 for B's 128x32), with
// constant-trip-count loops and data-only clamping -> no exec divergence,
// uniform ASYNCcnt watermark across all waves.
// ---------------------------------------------------------------------------
__device__ __forceinline__ void stage_async(
    const float* __restrict__ A, int lda, int m0, int k0,
    const float* __restrict__ B, int ldb, int bmode, int n0, int N,
    float* Astg, float* Bstg, int tid)
{
    // A tile: 128 x 32 fp32 row-major = 1024 16B chunks, 4 per thread
#pragma unroll
    for (int i = 0; i < 4; ++i) {
        const int c = tid + i * 256;
        const int row = c >> 3, ko = (c & 7) * 4;
        async_ld16(A + (size_t)(m0 + row) * lda + k0 + ko, Astg + row * TK + ko);
    }
    if (bmode == 1) {
        // B^T given N-major [N x K]: stage as [128][32] fp32
#pragma unroll
        for (int i = 0; i < 4; ++i) {
            const int c = tid + i * 256;
            const int row = c >> 3, ko = (c & 7) * 4;
            int n = n0 + row; if (n > N - 1) n = N - 1;   // clamp (data only, no branch)
            async_ld16(B + (size_t)n * ldb + k0 + ko, Bstg + row * TK + ko);
        }
    } else {
        // B row-major [K x N]: stage as [32][128] fp32 (transposed in convert pass)
#pragma unroll
        for (int i = 0; i < 4; ++i) {
            const int c = tid + i * 256;
            const int row = c >> 5, no = (c & 31) * 4;
            int n = n0 + no; if (n > N - 4) n = N - 4;    // clamp (N%4==0 for bmode 0)
            async_ld16(B + (size_t)(k0 + row) * ldb + n, Bstg + row * TN + no);
        }
    }
}

// ---------------------------------------------------------------------------
// Generic bf16-WMMA GEMM: C[M,N] = epilogue(alpha * A[M,K] x B + bias)
//  A: fp32 row-major, row stride lda
//  B: bmode==0 -> fp32 [K x N] row-major | bmode==1 -> fp32 [N x K] row-major
//  epilogue: 0 = none, 1 = GELU, 2 = += resid
//  Requirements: M % 128 == 0, K % 32 == 0. N guarded (N % 4 == 0 when bmode==0).
//  Pipeline: double-buffered async fp32 staging -> bf16 convert -> 8x WMMA/tile.
//  Block: 256 threads = 8 waves in a 2x4 grid; wave tile 64x32 (acc[4][2]).
// ---------------------------------------------------------------------------
__global__ __launch_bounds__(256)
void gemm_bf16_wmma(const float* __restrict__ A, int lda,
                    const float* __restrict__ B, int ldb, int bmode,
                    const float* __restrict__ bias,
                    const float* __restrict__ resid,
                    float* __restrict__ C, int ldc,
                    int M, int N, int K, float alpha, int epilogue)
{
    extern __shared__ char smem[];
    __bf16* As       = (__bf16*)(smem + AS_OFF);    // 128x40 bf16 tile
    __bf16* Bs       = (__bf16*)(smem + BS_OFF);    // 128x40 bf16 tile (N-major)
    float*  AstgBase = (float*)(smem + ASTG_OFF);   // 2 x 4096 fp32 staging
    float*  BstgBase = (float*)(smem + BSTG_OFF);   // 2 x 4096 fp32 staging

    const int tid  = threadIdx.x;
    const int lane = tid & 31;
    const int wave = tid >> 5;
    const int half = lane >> 4;     // 0: lanes 0-15, 1: lanes 16-31
    const int idx  = lane & 15;

    const int m0 = blockIdx.y * TM;
    const int n0 = blockIdx.x * TN;
    const int wm = (wave >> 2) * 64;   // wave row offset inside block tile
    const int wn = (wave & 3) * 32;    // wave col offset inside block tile

    const v8f zf = {0.f, 0.f, 0.f, 0.f, 0.f, 0.f, 0.f, 0.f};
    v8f acc[4][2];
#pragma unroll
    for (int mi = 0; mi < 4; ++mi)
#pragma unroll
        for (int ni = 0; ni < 2; ++ni) acc[mi][ni] = zf;

    const int nk = K / TK;
    // prologue: async-stage tile 0 into buffer 0
    stage_async(A, lda, m0, 0, B, ldb, bmode, n0, N, AstgBase, BstgBase, tid);

    int buf = 0;
    for (int kt = 0; kt < nk; ++kt) {
        // overlap: issue next tile's async copies, then wait only for the
        // older buffer (8 newer ops stay in flight behind the compute).
        if (kt + 1 < nk) {
            stage_async(A, lda, m0, (kt + 1) * TK, B, ldb, bmode, n0, N,
                        AstgBase + (buf ^ 1) * (TM * TK),
                        BstgBase + (buf ^ 1) * (TN * TK), tid);
            async_wait8();
        } else {
            async_wait0();
        }
        __syncthreads();   // all waves' staging[buf] is now resident

        const float* Astg = AstgBase + buf * (TM * TK);
        const float* Bstg = BstgBase + buf * (TN * TK);

        // ---- convert fp32 staging -> bf16 tiles (As / Bs), transposing B if needed
        {
            const int row = tid >> 1, kb = (tid & 1) * 16;
            const float* s = &Astg[row * TK + kb];
            v8bf t0, t1;
#pragma unroll
            for (int j = 0; j < 8; ++j) t0[j] = (__bf16)s[j];
#pragma unroll
            for (int j = 0; j < 8; ++j) t1[j] = (__bf16)s[8 + j];
            *(v8bf*)&As[row * LDSTR + kb]     = t0;
            *(v8bf*)&As[row * LDSTR + kb + 8] = t1;
        }
        if (bmode == 1) {
            const int row = tid >> 1, kb = (tid & 1) * 16;
            v8bf t0, t1;
            if (n0 + row < N) {
                const float* s = &Bstg[row * TK + kb];
#pragma unroll
                for (int j = 0; j < 8; ++j) t0[j] = (__bf16)s[j];
#pragma unroll
                for (int j = 0; j < 8; ++j) t1[j] = (__bf16)s[8 + j];
            } else {
#pragma unroll
                for (int j = 0; j < 8; ++j) { t0[j] = (__bf16)0.f; t1[j] = (__bf16)0.f; }
            }
            *(v8bf*)&Bs[row * LDSTR + kb]     = t0;
            *(v8bf*)&Bs[row * LDSTR + kb + 8] = t1;
        } else {
            const int n = tid & 127, kh = tid >> 7;
            v8bf t0, t1;
            if (n0 + n < N) {
                const float* s = &Bstg[kh * 16 * TN + n];  // stride-TN k walk (bank-clean)
#pragma unroll
                for (int j = 0; j < 8; ++j) t0[j] = (__bf16)s[(size_t)j * TN];
#pragma unroll
                for (int j = 0; j < 8; ++j) t1[j] = (__bf16)s[(size_t)(8 + j) * TN];
            } else {
#pragma unroll
                for (int j = 0; j < 8; ++j) { t0[j] = (__bf16)0.f; t1[j] = (__bf16)0.f; }
            }
            *(v8bf*)&Bs[n * LDSTR + kh * 16]     = t0;
            *(v8bf*)&Bs[n * LDSTR + kh * 16 + 8] = t1;
        }
        __syncthreads();   // bf16 tiles ready; also fences staging[buf] reads before its refill

        // ---- fragments per the ISA 16-bit layout:
        // lanes 0-15: K = kb..kb+7 in VGPR0-3, K = kb+16..kb+23 in VGPR4-7 (kb=0)
        // lanes 16-31: same with kb=8
        const int kb = half * 8;
        Frag afr[4], bfr[2];
#pragma unroll
        for (int i = 0; i < 4; ++i) {
            const int arow = wm + i * 16 + idx;
            afr[i].h[0] = *(const v8bf*)&As[arow * LDSTR + kb];
            afr[i].h[1] = *(const v8bf*)&As[arow * LDSTR + kb + 16];
        }
#pragma unroll
        for (int i = 0; i < 2; ++i) {
            const int brow = wn + i * 16 + idx;
            bfr[i].h[0] = *(const v8bf*)&Bs[brow * LDSTR + kb];
            bfr[i].h[1] = *(const v8bf*)&Bs[brow * LDSTR + kb + 16];
        }
#pragma unroll
        for (int mi = 0; mi < 4; ++mi)
#pragma unroll
            for (int ni = 0; ni < 2; ++ni)
                acc[mi][ni] = __builtin_amdgcn_wmma_f32_16x16x32_bf16(
                    false, afr[mi].v, false, bfr[ni].v,
                    (short)0, acc[mi][ni], false, false);
        buf ^= 1;
    }

    // ---- epilogue + store (C/D layout: vgpr v -> M=v(+8 for upper lanes), lane -> N)
#pragma unroll
    for (int mi = 0; mi < 4; ++mi) {
#pragma unroll
        for (int ni = 0; ni < 2; ++ni) {
            const int n = n0 + wn + ni * 16 + idx;
            if (n >= N) continue;
            const float bn = bias ? bias[n] : 0.f;
#pragma unroll
            for (int v = 0; v < 8; ++v) {
                const int m = m0 + wm + mi * 16 + (half ? v + 8 : v);
                float val = alpha * acc[mi][ni][v] + bn;
                const size_t off = (size_t)m * ldc + n;
                if (epilogue == 1)      val = gelu_tanh(val);
                else if (epilogue == 2) val += resid[off];
                C[off] = val;
            }
        }
    }
}

// ---------------------------------------------------------------------------
// h[row] = wte[ids[row]] + wpe[row % T]
// ---------------------------------------------------------------------------
__global__ __launch_bounds__(256)
void embed_kernel(const int* __restrict__ ids, const float* __restrict__ wte,
                  const float* __restrict__ wpe, float* __restrict__ h,
                  int T, int D)
{
    const int row = blockIdx.x;
    const int t = row % T;
    const int id = ids[row];
    const float* we = wte + (size_t)id * D;
    const float* wp = wpe + (size_t)t * D;
    float* o = h + (size_t)row * D;
    for (int d = threadIdx.x; d < D; d += 256) o[d] = we[d] + wp[d];
}

// ---------------------------------------------------------------------------
// LayerNorm per row (eps = 1e-5)
// ---------------------------------------------------------------------------
__global__ __launch_bounds__(256)
void ln_kernel(const float* __restrict__ in, const float* __restrict__ g,
               const float* __restrict__ b, float* __restrict__ out, int D)
{
    const int row = blockIdx.x;
    const float* x = in + (size_t)row * D;
    __shared__ float red[8];
    __shared__ float s_mean, s_rstd;
    const int lane = threadIdx.x & 31, wv = threadIdx.x >> 5;

    float s = 0.f;
    for (int d = threadIdx.x; d < D; d += 256) s += x[d];
    for (int o = 16; o > 0; o >>= 1) s += __shfl_down(s, o, 32);
    if (lane == 0) red[wv] = s;
    __syncthreads();
    if (threadIdx.x == 0) { float t = 0.f; for (int i = 0; i < 8; ++i) t += red[i]; s_mean = t / D; }
    __syncthreads();
    const float mean = s_mean;

    float v = 0.f;
    for (int d = threadIdx.x; d < D; d += 256) { float t = x[d] - mean; v += t * t; }
    for (int o = 16; o > 0; o >>= 1) v += __shfl_down(v, o, 32);
    __syncthreads();
    if (lane == 0) red[wv] = v;
    __syncthreads();
    if (threadIdx.x == 0) { float t = 0.f; for (int i = 0; i < 8; ++i) t += red[i]; s_rstd = rsqrtf(t / D + 1e-5f); }
    __syncthreads();
    const float rstd = s_rstd;

    float* o = out + (size_t)row * D;
    for (int d = threadIdx.x; d < D; d += 256)
        o[d] = (x[d] - mean) * rstd * g[d] + b[d];
}

// ---------------------------------------------------------------------------
// Causal softmax over scores[bh][i][:]; zeros j > i so the attn@V GEMM can
// contract over full K=T.
// ---------------------------------------------------------------------------
__global__ __launch_bounds__(256)
void softmax_kernel(float* __restrict__ sc, int T)
{
    const int gid = blockIdx.x;
    const int i = gid % T;
    float* row = sc + (size_t)gid * T;
    const int len = i + 1;
    __shared__ float red[8];
    __shared__ float s_m, s_s;
    const int lane = threadIdx.x & 31, wv = threadIdx.x >> 5;

    float mx = -3.0e38f;
    for (int j = threadIdx.x; j < len; j += 256) mx = fmaxf(mx, row[j]);
    for (int o = 16; o > 0; o >>= 1) mx = fmaxf(mx, __shfl_down(mx, o, 32));
    if (lane == 0) red[wv] = mx;
    __syncthreads();
    if (threadIdx.x == 0) { float t = red[0]; for (int k = 1; k < 8; ++k) t = fmaxf(t, red[k]); s_m = t; }
    __syncthreads();
    const float m = s_m;

    float sum = 0.f;
    for (int j = threadIdx.x; j < len; j += 256) { float e = expf(row[j] - m); row[j] = e; sum += e; }
    for (int o = 16; o > 0; o >>= 1) sum += __shfl_down(sum, o, 32);
    __syncthreads();
    if (lane == 0) red[wv] = sum;
    __syncthreads();
    if (threadIdx.x == 0) { float t = 0.f; for (int k = 0; k < 8; ++k) t += red[k]; s_s = t; }
    __syncthreads();
    const float inv = 1.f / s_s;

    for (int j = threadIdx.x; j < len; j += 256) row[j] *= inv;
    for (int j = len + (int)threadIdx.x; j < T; j += 256) row[j] = 0.f;
}

// ---------------------------------------------------------------------------
// Per-token MoE adapter: h += sum_e softmax(x@Wtask)_e * ((x @ A_e) @ B_e)
// D=768, E=6, R=24. One workgroup per token.
// ---------------------------------------------------------------------------
#define ADP_D 768
#define ADP_E 6
#define ADP_R 24

__global__ __launch_bounds__(256)
void adapter_kernel(const float* __restrict__ x, const float* __restrict__ Wtask,
                    const float* __restrict__ Aa, const float* __restrict__ Ba,
                    float* __restrict__ h)
{
    const int row = blockIdx.x;
    const float* xr = x + (size_t)row * ADP_D;
    __shared__ float xs[ADP_D];
    __shared__ float gred[ADP_E];
    __shared__ float gates[ADP_E];
    __shared__ float low[ADP_E * ADP_R];

    for (int d = threadIdx.x; d < ADP_D; d += 256) xs[d] = xr[d];
    __syncthreads();

    const int wave = threadIdx.x >> 5, lane = threadIdx.x & 31;
    if (wave < ADP_E) {
        float s = 0.f;
        for (int d = lane; d < ADP_D; d += 32) s += xs[d] * Wtask[(size_t)d * ADP_E + wave];
        for (int o = 16; o > 0; o >>= 1) s += __shfl_down(s, o, 32);
        if (lane == 0) gred[wave] = s;
    }
    __syncthreads();
    if (threadIdx.x == 0) {
        float mx = -3.0e38f;
        for (int e = 0; e < ADP_E; ++e) mx = fmaxf(mx, gred[e]);
        float ss = 0.f;
        for (int e = 0; e < ADP_E; ++e) { float v = expf(gred[e] - mx); gates[e] = v; ss += v; }
        for (int e = 0; e < ADP_E; ++e) gates[e] /= ss;
    }
    __syncthreads();

    if (threadIdx.x < ADP_E * ADP_R) {
        const int e = threadIdx.x / ADP_R, r = threadIdx.x % ADP_R;
        const float* Ac = Aa + (size_t)e * ADP_D * ADP_R + r;
        float s = 0.f;
        for (int d = 0; d < ADP_D; ++d) s += xs[d] * Ac[(size_t)d * ADP_R];
        low[threadIdx.x] = s * gates[e];
    }
    __syncthreads();

    for (int d = threadIdx.x; d < ADP_D; d += 256) {
        float acc = 0.f;
        for (int er = 0; er < ADP_E * ADP_R; ++er)
            acc += low[er] * Ba[(size_t)er * ADP_D + d];
        h[(size_t)row * ADP_D + d] += acc;
    }
}

// ---------------------------------------------------------------------------
// Host orchestration
// ---------------------------------------------------------------------------
static inline void gemm_launch(hipStream_t stream,
                               const float* A, int lda,
                               const float* B, int ldb, int bmode,
                               const float* bias, const float* resid,
                               float* C, int ldc,
                               int M, int N, int K, float alpha, int epilogue)
{
    dim3 grid((N + TN - 1) / TN, M / TM);
    gemm_bf16_wmma<<<grid, 256, SMEM_BYTES, stream>>>(A, lda, B, ldb, bmode, bias, resid,
                                                      C, ldc, M, N, K, alpha, epilogue);
}

extern "C" void kernel_launch(void* const* d_in, const int* in_sizes, int n_in,
                              void* d_out, int out_size, void* d_ws, size_t ws_size,
                              hipStream_t stream)
{
    (void)in_sizes; (void)n_in; (void)out_size; (void)ws_size;

    const int*   ids   = (const int*)d_in[0];
    const float* wte   = (const float*)d_in[1];
    const float* wpe   = (const float*)d_in[2];
    const float* ln1g  = (const float*)d_in[3];
    const float* ln1b  = (const float*)d_in[4];
    const float* Wqkv  = (const float*)d_in[5];
    const float* bqkv  = (const float*)d_in[6];
    const float* Wo    = (const float*)d_in[7];
    const float* bo    = (const float*)d_in[8];
    const float* ln2g  = (const float*)d_in[9];
    const float* ln2b  = (const float*)d_in[10];
    const float* Wfc   = (const float*)d_in[11];
    const float* bfc   = (const float*)d_in[12];
    const float* Wproj = (const float*)d_in[13];
    const float* bproj = (const float*)d_in[14];
    const float* lnfg  = (const float*)d_in[15];
    const float* lnfb  = (const float*)d_in[16];
    const float* Aad   = (const float*)d_in[17];
    const float* Bad   = (const float*)d_in[18];
    const float* Wtask = (const float*)d_in[19];
    float* out = (float*)d_out;

    const int Bn = 2, T = 1024, D = 768, Hh = 12, DH = 64, L = 12, V = 50257;
    const int NT = Bn * T;

    float* ws  = (float*)d_ws;
    float* h   = ws;
    float* xb  = h   + (size_t)NT * D;
    float* qkv = xb  + (size_t)NT * D;
    float* sc  = qkv + (size_t)NT * 3 * D;
    float* ctx = sc  + (size_t)Bn * Hh * T * T;
    float* fcb = ctx + (size_t)NT * D;

    embed_kernel<<<NT, 256, 0, stream>>>(ids, wte, wpe, h, T, D);

    for (int l = 0; l < L; ++l) {
        // x = LN1(h)
        ln_kernel<<<NT, 256, 0, stream>>>(h, ln1g + (size_t)l * D, ln1b + (size_t)l * D, xb, D);
        // qkv = x @ Wqkv + bqkv
        gemm_launch(stream, xb, D, Wqkv + (size_t)l * D * 3 * D, 3 * D, 0,
                    bqkv + (size_t)l * 3 * D, nullptr, qkv, 3 * D,
                    NT, 3 * D, D, 1.f, 0);
        // scores = q @ k^T / sqrt(DH), per (b, head)
        for (int b = 0; b < Bn; ++b)
            for (int hh = 0; hh < Hh; ++hh) {
                const float* q = qkv + (size_t)b * T * 3 * D + hh * DH;
                const float* k = q + D;
                float* s = sc + ((size_t)(b * Hh + hh)) * T * T;
                gemm_launch(stream, q, 3 * D, k, 3 * D, 1, nullptr, nullptr,
                            s, T, T, T, DH, 0.125f, 0);
            }
        // causal softmax (zeros upper triangle)
        softmax_kernel<<<Bn * Hh * T, 256, 0, stream>>>(sc, T);
        // ctx = attn @ v, scattered back to [B,T,H*DH]
        for (int b = 0; b < Bn; ++b)
            for (int hh = 0; hh < Hh; ++hh) {
                const float* s = sc + ((size_t)(b * Hh + hh)) * T * T;
                const float* v = qkv + (size_t)b * T * 3 * D + 2 * D + hh * DH;
                gemm_launch(stream, s, T, v, 3 * D, 0, nullptr, nullptr,
                            ctx + (size_t)b * T * D + hh * DH, D,
                            T, DH, T, 1.f, 0);
            }
        // h = h + ctx @ Wo + bo
        gemm_launch(stream, ctx, D, Wo + (size_t)l * D * D, D, 0,
                    bo + (size_t)l * D, h, h, D, NT, D, D, 1.f, 2);
        // MoE adapter on layers 0,2,4,6 (adds into h)
        const int ai = (l == 0) ? 0 : (l == 2) ? 1 : (l == 4) ? 2 : (l == 6) ? 3 : -1;
        if (ai >= 0)
            adapter_kernel<<<NT, 256, 0, stream>>>(
                xb, Wtask,
                Aad + (size_t)ai * ADP_E * ADP_D * ADP_R,
                Bad + (size_t)ai * ADP_E * ADP_R * ADP_D, h);
        // x = LN2(h)
        ln_kernel<<<NT, 256, 0, stream>>>(h, ln2g + (size_t)l * D, ln2b + (size_t)l * D, xb, D);
        // fcb = gelu(x @ Wfc + bfc)
        gemm_launch(stream, xb, D, Wfc + (size_t)l * D * 4 * D, 4 * D, 0,
                    bfc + (size_t)l * 4 * D, nullptr, fcb, 4 * D,
                    NT, 4 * D, D, 1.f, 1);
        // h = h + fcb @ Wproj + bproj
        gemm_launch(stream, fcb, 4 * D, Wproj + (size_t)l * 4 * D * D, D, 0,
                    bproj + (size_t)l * D, h, h, D, NT, D, 4 * D, 1.f, 2);
    }

    // final LN + tied lm_head: logits = LNf(h) @ wte^T  (wte is [V,D] = N-major B)
    ln_kernel<<<NT, 256, 0, stream>>>(h, lnfg, lnfb, xb, D);
    gemm_launch(stream, xb, D, wte, D, 1, nullptr, nullptr,
                out, V, NT, V, D, 1.f, 0);
}
